// PNA_Net_41618233099025
// MI455X (gfx1250) — compile-verified, matrix operand order checked by reading, not run on hardware
//
#include <hip/hip_runtime.h>

// ---------------- problem constants ----------------
#define N_    10000
#define E_    160000
#define F_    40
#define T_    5
#define EDIM_ 16
#define HID_  256
#define OUT_  128
#define NCLS_ 10
#define G_    100
#define KP    128   // 3F=120 padded to 128 (4 k-tiles of 32)
#define NC    200   // T*F output channels
#define NP    208   // padded to 13 n-tiles of 16
#define KT    4
#define NT    13

typedef __attribute__((ext_vector_type(16))) __bf16 v16bf;
typedef __attribute__((ext_vector_type(8)))  float  v8f;

#define WMMA_BF16(a, b, c) \
  __builtin_amdgcn_wmma_f32_16x16x32_bf16(false, (a), false, (b), (short)0, (c), false, false)

// ---------------- helpers ----------------
__device__ inline unsigned short f2bf(float f) {
  unsigned u = __float_as_uint(f);
  unsigned r = u + 0x7FFFu + ((u >> 16) & 1u);   // round-to-nearest-even
  return (unsigned short)(r >> 16);
}

// ---------------- init kernels ----------------
__global__ void k_fill_f(float* p, float v, int n) {
  int i = blockIdx.x * blockDim.x + threadIdx.x;
  if (i < n) p[i] = v;
}
__global__ void k_fill_u(unsigned* p, unsigned v, int n) {
  int i = blockIdx.x * blockDim.x + threadIdx.x;
  if (i < n) p[i] = v;
}

// ---------------- degree count ----------------
__global__ void k_count(const int* dst, int* cnt) {
  int e = blockIdx.x * blockDim.x + threadIdx.x;
  if (e < E_) atomicAdd(&cnt[dst[e]], 1);
}

// ---------------- exclusive scan of cnt -> rowptr/cursor, plus avg_log ----------------
__global__ void __launch_bounds__(256)
k_scan(const int* __restrict__ cnt, int* __restrict__ rowptr,
       int* __restrict__ cursor, float* __restrict__ avga) {
  __shared__ int   part[256];
  __shared__ float lpart[256];
  const int CH = (N_ + 255) / 256;     // 40 nodes per thread
  int tid = threadIdx.x;
  int base = tid * CH;
  int s = 0; float ls = 0.0f;
  for (int i = 0; i < CH; ++i) {
    int idx = base + i;
    if (idx < N_) { int c = cnt[idx]; s += c; ls += logf((float)c + 1.0f); }
  }
  part[tid] = s; lpart[tid] = ls;
  __syncthreads();
  if (tid == 0) {
    int acc = 0; float lacc = 0.0f;
    for (int i = 0; i < 256; ++i) { int t = part[i]; part[i] = acc; acc += t; lacc += lpart[i]; }
    rowptr[N_] = acc;                  // == E_
    avga[0] = lacc / (float)N_;        // avg_log
  }
  __syncthreads();
  int acc = part[tid];
  for (int i = 0; i < CH; ++i) {
    int idx = base + i;
    if (idx < N_) { rowptr[idx] = acc; cursor[idx] = acc; acc += cnt[idx]; }
  }
}

// ---------------- scatter edge ids into CSR slots ----------------
__global__ void k_scatter(const int* __restrict__ dst, int* __restrict__ cursor,
                          int* __restrict__ perm) {
  int e = blockIdx.x * blockDim.x + threadIdx.x;
  if (e < E_) {
    int s = atomicAdd(&cursor[dst[e]], 1);
    perm[s] = e;
  }
}

// ---------------- per-layer: pack pre_w into WMMA B-fragment order ----------------
// wpack: [nt][kt][lane][idx] bf16; lane = n%16 + 16*((kin>>3)&1), idx = (kin&7) + 8*(kin>>4)
__global__ void k_pack_w(const float* __restrict__ pre_w, const float* __restrict__ pre_b,
                         unsigned short* __restrict__ wpack, float* __restrict__ biasb) {
  int tid = blockIdx.x * blockDim.x + threadIdx.x;   // NT*KT*32*16 = 26624 threads
  if (tid >= NT * KT * 32 * 16) return;
  int idx  = tid & 15;
  int lane = (tid >> 4) & 31;
  int kt   = (tid >> 9) & 3;
  int nt   = tid >> 11;
  int khalf = lane >> 4;
  int kin = (idx & 7) | (khalf << 3) | ((idx >> 3) << 4);
  int k = kt * 32 + kin;
  int n = nt * 16 + (lane & 15);
  float v = 0.0f;
  if (k < 3 * F_ && n < NC) {
    int t = n / F_, o = n % F_;
    v = pre_w[(t * (3 * F_) + k) * F_ + o];          // pre_w: [T, 3F, F]
  }
  wpack[tid] = f2bf(v);
  if (tid < NP) biasb[tid] = (tid < NC) ? pre_b[tid] : 0.0f;
}

// ---------------- build h = [x_dst | x_src | edge_lin] in A-fragment order, CSR slot order ----------------
__global__ void k_build_h(const float* __restrict__ x, const float* __restrict__ eattr,
                          const float* __restrict__ ew, const float* __restrict__ eb,
                          const int* __restrict__ src, const int* __restrict__ dst,
                          const int* __restrict__ perm, unsigned short* __restrict__ hpack) {
  int tid = blockIdx.x * blockDim.x + threadIdx.x;   // E_*KP threads
  int slot = tid >> 7;
  int k = tid & 127;
  if (slot >= E_) return;
  int e = perm[slot];
  float v = 0.0f;
  if (k < F_) {
    v = x[dst[e] * F_ + k];
  } else if (k < 2 * F_) {
    v = x[src[e] * F_ + (k - F_)];
  } else if (k < 3 * F_) {
    int j = k - 2 * F_;
    float a = eb[j];
    const float* ea = eattr + (long)e * EDIM_;
    #pragma unroll
    for (int i = 0; i < EDIM_; ++i) a += ea[i] * ew[i * F_ + j];
    v = a;
  }
  int mtile = slot >> 4, m = slot & 15;
  int kt = k >> 5, kin = k & 31;
  int lane = m + 16 * ((kin >> 3) & 1);
  int idx  = (kin & 7) + 8 * (kin >> 4);
  hpack[(((mtile * KT) + kt) * 32 + lane) * 16 + idx] = f2bf(v);
}

// ---------------- WMMA msg GEMM: 2 M-tiles/wave, A in registers, loop 13 B tiles ----------------
// msgT layout: [channel 0..199][slot 0..E-1]  -> each lane stores 8 contiguous floats per tile
__global__ void __launch_bounds__(32)
k_msg_wmma(const unsigned short* __restrict__ hpack, const unsigned short* __restrict__ wpack,
           const float* __restrict__ biasb, float* __restrict__ msgT) {
  int mt0  = blockIdx.x * 2;
  int mt1  = mt0 + 1;
  int lane = threadIdx.x;
  const v16bf* A = (const v16bf*)hpack;
  const v16bf* B = (const v16bf*)wpack;
  // 8 A fragments held in VGPRs for the whole kernel (two 16x128 tiles)
  v16bf a0 = A[(mt0 * KT + 0) * 32 + lane];
  v16bf a1 = A[(mt0 * KT + 1) * 32 + lane];
  v16bf a2 = A[(mt0 * KT + 2) * 32 + lane];
  v16bf a3 = A[(mt0 * KT + 3) * 32 + lane];
  v16bf a4 = A[(mt1 * KT + 0) * 32 + lane];
  v16bf a5 = A[(mt1 * KT + 1) * 32 + lane];
  v16bf a6 = A[(mt1 * KT + 2) * 32 + lane];
  v16bf a7 = A[(mt1 * KT + 3) * 32 + lane];
  int col_lo = lane & 15;
  int rb0 = mt0 * 16 + 8 * (lane >> 4);
  int rb1 = mt1 * 16 + 8 * (lane >> 4);
  #pragma unroll 1
  for (int nt = 0; nt < NT; ++nt) {
    // batch all 4 B-fragment loads (8x b128) before first use
    v16bf b0 = B[(nt * KT + 0) * 32 + lane];
    v16bf b1 = B[(nt * KT + 1) * 32 + lane];
    v16bf b2 = B[(nt * KT + 2) * 32 + lane];
    v16bf b3 = B[(nt * KT + 3) * 32 + lane];
    v8f acc0 = {}, acc1 = {};
    // two independent accumulator chains interleaved on shared B
    acc0 = WMMA_BF16(a0, b0, acc0);  acc1 = WMMA_BF16(a4, b0, acc1);
    acc0 = WMMA_BF16(a1, b1, acc0);  acc1 = WMMA_BF16(a5, b1, acc1);
    acc0 = WMMA_BF16(a2, b2, acc0);  acc1 = WMMA_BF16(a6, b2, acc1);
    acc0 = WMMA_BF16(a3, b3, acc0);  acc1 = WMMA_BF16(a7, b3, acc1);
    int col = nt * 16 + col_lo;
    if (col < NC) {
      float bias = biasb[col];
      v8f o0, o1;
      #pragma unroll
      for (int j = 0; j < 8; ++j) { o0[j] = acc0[j] + bias; o1[j] = acc1[j] + bias; }
      float* base = msgT + (long)col * E_;
      *(v8f*)(base + rb0) = o0;   // 32B contiguous stores
      *(v8f*)(base + rb1) = o1;
    }
  }
}

// ---------------- fused CSR aggregation + post-tower matvec + lin + BN stats ----------------
__global__ void __launch_bounds__(256)
k_aggpost(const float* __restrict__ xin, const int* __restrict__ rowptr,
          const float* __restrict__ avga, const float* __restrict__ msgT,
          const float* __restrict__ post_w, const float* __restrict__ post_b,
          const float* __restrict__ lin_w, const float* __restrict__ lin_b,
          float* __restrict__ y, float* __restrict__ bnsum, float* __restrict__ bnsumsq) {
  __shared__ float sh[T_ * 520];
  __shared__ float out40[F_];
  int n = blockIdx.x, tid = threadIdx.x;
  int r0 = rowptr[n], r1 = rowptr[n + 1];
  float cntv = (float)(r1 - r0);
  float c  = fmaxf(cntv, 1.0f);
  float lg = logf(c + 1.0f);
  float avg = avga[0];
  float s2 = lg / avg, s3 = avg / lg;
  if (tid < F_) {
    float xv = xin[n * F_ + tid];
    #pragma unroll
    for (int t = 0; t < T_; ++t) sh[t * 520 + tid] = xv;
  }
  if (tid < NC) {
    const float* mrow = msgT + (long)tid * E_;
    float sm = 0.0f, sq = 0.0f, mn = 3.402823466e+38f, mx = -3.402823466e+38f;
    for (int s = r0; s < r1; ++s) {
      float v = mrow[s];
      sm += v; sq += v * v;
      mn = fminf(mn, v); mx = fmaxf(mx, v);
    }
    if (r1 <= r0) { mn = 0.0f; mx = 0.0f; }
    float mean = sm / c;
    float sd = sqrtf(fmaxf(sq / c - mean * mean, 0.0f) + 1e-5f);
    int t = tid / F_, f = tid % F_;
    int b = t * 520 + F_;
    sh[b + 0   + f] = mean;      sh[b + 40  + f] = mn;
    sh[b + 80  + f] = mx;        sh[b + 120 + f] = sd;
    sh[b + 160 + f] = mean * s2; sh[b + 200 + f] = mn * s2;
    sh[b + 240 + f] = mx * s2;   sh[b + 280 + f] = sd * s2;
    sh[b + 320 + f] = mean * s3; sh[b + 360 + f] = mn * s3;
    sh[b + 400 + f] = mx * s3;   sh[b + 440 + f] = sd * s3;
  }
  __syncthreads();
  if (tid < F_) {                                // 40 outputs: tower t, channel oo
    int t = tid >> 3, oo = tid & 7;
    float a = post_b[t * 8 + oo];
    const float* wp = post_w + (long)(t * 520) * 8 + oo;   // post_w: [T,520,8]
    const float* s  = sh + t * 520;
    for (int i = 0; i < 520; ++i) a += s[i] * wp[i * 8];
    out40[tid] = a;
  }
  __syncthreads();
  if (tid < F_) {
    float a = lin_b[tid];
    #pragma unroll 8
    for (int i = 0; i < F_; ++i) a += out40[i] * lin_w[i * F_ + tid];
    y[n * F_ + tid] = a;
    atomicAdd(&bnsum[tid], a);
    atomicAdd(&bnsumsq[tid], a * a);
  }
}

// ---------------- batch-norm + relu ----------------
__global__ void k_bn(const float* __restrict__ y, const float* __restrict__ bnsum,
                     const float* __restrict__ bnsumsq, const float* __restrict__ g,
                     const float* __restrict__ b, float* __restrict__ xout) {
  int i = blockIdx.x * blockDim.x + threadIdx.x;
  if (i >= N_ * F_) return;
  int ch = i % F_;
  float m = bnsum[ch] / (float)N_;
  float var = bnsumsq[ch] / (float)N_ - m * m;
  float v = (y[i] - m) * rsqrtf(var + 1e-5f) * g[ch] + b[ch];
  xout[i] = fmaxf(v, 0.0f);
}

// ---------------- final MLP + global add pool ----------------
__global__ void __launch_bounds__(256)
k_mlp(const float* __restrict__ h,
      const float* __restrict__ w1, const float* __restrict__ b1,
      const float* __restrict__ w2, const float* __restrict__ b2,
      const float* __restrict__ w3, const float* __restrict__ b3,
      const int* __restrict__ batch, float* __restrict__ out) {
  __shared__ float xl[F_];
  __shared__ float hh[HID_];
  __shared__ float o2[OUT_];
  int n = blockIdx.x, tid = threadIdx.x;
  if (tid < F_) xl[tid] = h[n * F_ + tid];
  __syncthreads();
  {
    float a = b1[tid];
    #pragma unroll 8
    for (int i = 0; i < F_; ++i) a += xl[i] * w1[i * HID_ + tid];
    hh[tid] = fmaxf(a, 0.0f);
  }
  __syncthreads();
  if (tid < OUT_) {
    float a = b2[tid];
    #pragma unroll 8
    for (int i = 0; i < HID_; ++i) a += hh[i] * w2[i * OUT_ + tid];
    o2[tid] = fmaxf(a, 0.0f);
  }
  __syncthreads();
  if (tid < NCLS_) {
    float a = b3[tid];
    #pragma unroll 8
    for (int i = 0; i < OUT_; ++i) a += o2[i] * w3[i * NCLS_ + tid];
    atomicAdd(&out[batch[n] * NCLS_ + tid], a);
  }
}

// ---------------- host ----------------
extern "C" void kernel_launch(void* const* d_in, const int* in_sizes, int n_in,
                              void* d_out, int out_size, void* d_ws, size_t ws_size,
                              hipStream_t stream) {
  (void)in_sizes; (void)n_in; (void)out_size; (void)ws_size;

  // input order: x, edge_attr, [params flattened], edge_index, batch
  // params: layers[0..1] each {bn_b,bn_g,edge_b,edge_w,lin_b,lin_w,post_b,post_w,pre_b,pre_w},
  //         then lin1(w,b), lin2(w,b), lin3(w,b)
  const float* x_in  = (const float*)d_in[0];
  const float* eattr = (const float*)d_in[1];
  const int lay_base[2] = {2, 12};
  const float* lin1w = (const float*)d_in[22];
  const float* lin1b = (const float*)d_in[23];
  const float* lin2w = (const float*)d_in[24];
  const float* lin2b = (const float*)d_in[25];
  const float* lin3w = (const float*)d_in[26];
  const float* lin3b = (const float*)d_in[27];
  const int* eidx  = (const int*)d_in[28];
  const int* batch = (const int*)d_in[29];
  const int* src = eidx;        // edge_index[0]
  const int* dst = eidx + E_;   // edge_index[1]
  float* out = (float*)d_out;

  // workspace layout
  char* p = (char*)d_ws;
  auto alloc = [&](size_t bytes) -> void* {
    void* r = (void*)p;
    p += (bytes + 255) & ~(size_t)255;
    return r;
  };
  unsigned short* hpack = (unsigned short*)alloc((size_t)E_ * KP * 2);        // 41 MB
  float*    msgT  = (float*)alloc((size_t)NC * E_ * 4);                       // 128 MB
  unsigned short* wpack = (unsigned short*)alloc((size_t)NT * KT * 32 * 16 * 2);
  float*    biasb = (float*)alloc(NP * 4);
  int*      cnt   = (int*)alloc((size_t)N_ * 4);
  int*      rowptr= (int*)alloc((size_t)(N_ + 1) * 4);
  int*      cursor= (int*)alloc((size_t)N_ * 4);
  int*      perm  = (int*)alloc((size_t)E_ * 4);
  float*    avga  = (float*)alloc(256);
  float*    ybuf  = (float*)alloc((size_t)N_ * F_ * 4);
  float*    xb1   = (float*)alloc((size_t)N_ * F_ * 4);
  float*    xb2   = (float*)alloc((size_t)N_ * F_ * 4);
  float*    bnsum = (float*)alloc(256);
  float*    bnsq  = (float*)alloc(256);

  const int B = 256;
  // ---- graph structure (layer-invariant): degree, CSR rowptr, slot permutation, avg_log ----
  k_fill_u<<<(N_ + B - 1) / B, B, 0, stream>>>((unsigned*)cnt, 0u, N_);
  k_count<<<(E_ + B - 1) / B, B, 0, stream>>>(dst, cnt);
  k_scan<<<1, 256, 0, stream>>>(cnt, rowptr, cursor, avga);
  k_scatter<<<(E_ + B - 1) / B, B, 0, stream>>>(dst, cursor, perm);
  k_fill_f<<<(G_ * NCLS_ + B - 1) / B, B, 0, stream>>>(out, 0.0f, G_ * NCLS_);

  const float* xcur = x_in;
  for (int l = 0; l < 2; ++l) {
    int bi = lay_base[l];
    const float* bn_b   = (const float*)d_in[bi + 0];
    const float* bn_g   = (const float*)d_in[bi + 1];
    const float* edge_b = (const float*)d_in[bi + 2];
    const float* edge_w = (const float*)d_in[bi + 3];
    const float* lin_b  = (const float*)d_in[bi + 4];
    const float* lin_w  = (const float*)d_in[bi + 5];
    const float* post_b = (const float*)d_in[bi + 6];
    const float* post_w = (const float*)d_in[bi + 7];
    const float* pre_b  = (const float*)d_in[bi + 8];
    const float* pre_w  = (const float*)d_in[bi + 9];

    k_fill_f<<<1, 64, 0, stream>>>(bnsum, 0.0f, 64);
    k_fill_f<<<1, 64, 0, stream>>>(bnsq, 0.0f, 64);

    k_pack_w<<<(NT * KT * 32 * 16 + B - 1) / B, B, 0, stream>>>(pre_w, pre_b, wpack, biasb);
    k_build_h<<<(E_ * KP) / B, B, 0, stream>>>(xcur, eattr, edge_w, edge_b, src, dst, perm, hpack);
    k_msg_wmma<<<E_ / 32, 32, 0, stream>>>(hpack, wpack, biasb, msgT);
    k_aggpost<<<N_, 256, 0, stream>>>(xcur, rowptr, avga, msgT,
                                      post_w, post_b, lin_w, lin_b, ybuf, bnsum, bnsq);
    float* xnext = (l == 0) ? xb1 : xb2;
    k_bn<<<(N_ * F_ + B - 1) / B, B, 0, stream>>>(ybuf, bnsum, bnsq, bn_g, bn_b, xnext);
    xcur = xnext;
  }

  k_mlp<<<N_, 256, 0, stream>>>(xcur, lin1w, lin1b, lin2w, lin2b, lin3w, lin3b, batch, out);
}
